// GraphConvNNEdge_83854941487725
// MI455X (gfx1250) — matrix operand order repaired and checked by reading, not compiled, not run on hardware
//
#include <hip/hip_runtime.h>
#include <hip/hip_bf16.h>

typedef __attribute__((ext_vector_type(16))) _Float16 v16h;
typedef __attribute__((ext_vector_type(8)))  float    v8f;

// ---------------------------------------------------------------------------
// Utility: f32 -> f16 conversion (grid-stride)
// ---------------------------------------------------------------------------
__global__ void k_f32_to_f16(const float* __restrict__ in, _Float16* __restrict__ out, int n) {
    for (int i = blockIdx.x * blockDim.x + threadIdx.x; i < n; i += gridDim.x * blockDim.x)
        out[i] = (_Float16)in[i];
}

// ---------------------------------------------------------------------------
// Degree / normalization (GCN with self loops: deg >= 1 always)
// ---------------------------------------------------------------------------
__global__ void k_deg_init(unsigned* __restrict__ deg, int n) {
    int i = blockIdx.x * blockDim.x + threadIdx.x;
    if (i < n) deg[i] = 1u;   // self loop
}

__global__ void k_deg_count(const int* __restrict__ edges, unsigned* __restrict__ deg, int E) {
    int e = blockIdx.x * blockDim.x + threadIdx.x;
    if (e < E) atomicAdd(&deg[edges[2 * e + 1]], 1u);
}

__global__ void k_dinv(const unsigned* __restrict__ deg, float* __restrict__ dinv, int n) {
    int i = blockIdx.x * blockDim.x + threadIdx.x;
    if (i < n) dinv[i] = rsqrtf((float)deg[i]);
}

// ---------------------------------------------------------------------------
// WMMA GEMM:  out[N,64] = concat(xh[N,32], hprev[N,64])[ : , 0:KDIM] @ W[KDIM,64]
// One wave per 16-row M tile; 4 f32 accumulators cover the 64 output columns.
// All epilogue variants are compile-time specialized (no runtime flag
// branches): DINV scales by dinv[row] (source-side GCN norm), BIAS/RELU for
// the MLP layer, AGG mirrors into the aggregation buffer (self-loop init),
// HOUT writes the f16 copy.
// ---------------------------------------------------------------------------
template <int KDIM, bool DINV, bool BIAS, bool RELU, bool AGG, bool HOUT>
__global__ __launch_bounds__(256) void k_gcn_gemm(
        const _Float16* __restrict__ xh,     // [N,32]
        const _Float16* __restrict__ hprev,  // [N,64] (unused when KDIM==32)
        const _Float16* __restrict__ W,      // [KDIM,64] row-major
        const float*    __restrict__ bias,   // [64]   (BIAS)
        const float*    __restrict__ dinv,   // [N]    (DINV)
        float*          __restrict__ ys,     // [N,64]
        float*          __restrict__ agg,    // [N,64] (AGG)
        _Float16*       __restrict__ hout,   // [N,64] (HOUT)
        int Nn, int Ntiles)
{
    constexpr int KC = KDIM / 32;
    const int wave = threadIdx.x >> 5;
    const int lane = threadIdx.x & 31;
    const int ln   = lane & 15;      // N-index within tile (B/C/D), M-index (A)
    const int lh   = lane >> 4;      // half-wave selector

    // --- B fragments: built once per wave, reused over all M tiles ---------
    // Layout (16-bit B 32x16, this code's contiguous k order, not the ISA table): VGPR j holds rows K = lh*16 + 2j,
    // 2j+1 at column N = lane%16.
    v16h bfrag[KC][4];
#pragma unroll
    for (int kc = 0; kc < KC; ++kc)
#pragma unroll
        for (int nt = 0; nt < 4; ++nt) {
            const int n = nt * 16 + ln;
#pragma unroll
            for (int j = 0; j < 8; ++j) {
                const int k = kc * 32 + lh * 16 + 2 * j;
                bfrag[kc][nt][2 * j]     = W[k * 64 + n];
                bfrag[kc][nt][2 * j + 1] = W[(k + 1) * 64 + n];
            }
        }

    // bias per output column (uniform over tiles) -- hoisted
    float bv[4] = {0.f, 0.f, 0.f, 0.f};
    if constexpr (BIAS) {
#pragma unroll
        for (int nt = 0; nt < 4; ++nt) bv[nt] = bias[nt * 16 + ln];
    }

    for (int tile = blockIdx.x * 8 + wave; tile < Ntiles; tile += gridDim.x * 8) {
        int mrow = tile * 16 + ln;
        int mr   = mrow < Nn ? mrow : Nn - 1;   // clamp (N%16==0 in practice)

        v8f acc[4] = {v8f{}, v8f{}, v8f{}, v8f{}};

#pragma unroll
        for (int kc = 0; kc < KC; ++kc) {
            // A fragment (16-bit A 16x32): VGPR j holds
            // K = lh*16 + 2j (+1) for row M = lane%16 (matches B).
            // (compiler merges these into two global_load_b128 per chunk)
            v16h afrag;
#pragma unroll
            for (int j = 0; j < 8; ++j) {
                const int k = kc * 32 + lh * 16 + 2 * j;
                const _Float16* src = (k < 32) ? (xh + mr * 32 + k)
                                               : (hprev + mr * 64 + (k - 32));
                afrag[2 * j]     = src[0];
                afrag[2 * j + 1] = src[1];
            }
#pragma unroll
            for (int nt = 0; nt < 4; ++nt)
                acc[nt] = __builtin_amdgcn_wmma_f32_16x16x32_f16(
                    false, afrag, false, bfrag[kc][nt], (short)0, acc[nt],
                    false, false);
        }

        // --- dinv for the 8 accumulator rows: contiguous -> 2x b128 loads --
        float dv[8];
        if constexpr (DINV) {
            const int rbase = tile * 16 + lh * 8;
            const int rb = (rbase + 8 <= Nn) ? rbase : 0;   // uniform clamp
            const float4 d0 = ((const float4*)(dinv + rb))[0];
            const float4 d1 = ((const float4*)(dinv + rb))[1];
            dv[0] = d0.x; dv[1] = d0.y; dv[2] = d0.z; dv[3] = d0.w;
            dv[4] = d1.x; dv[5] = d1.y; dv[6] = d1.z; dv[7] = d1.w;
        }

        // --- Epilogue: D layout row = tile*16 + lh*8 + r, col = nt*16 + ln -
        const bool fullTile = ((tile + 1) * 16 <= Nn);
        if (fullTile) {
#pragma unroll
            for (int nt = 0; nt < 4; ++nt) {
                const int col = nt * 16 + ln;
#pragma unroll
                for (int r = 0; r < 8; ++r) {
                    const int row = tile * 16 + lh * 8 + r;
                    float v = acc[nt][r];
                    if constexpr (DINV) v *= dv[r];
                    if constexpr (BIAS) v += bv[nt];
                    if constexpr (RELU) v = fmaxf(v, 0.0f);
                    ys[row * 64 + col] = v;
                    if constexpr (AGG)  agg[row * 64 + col] = v;
                    if constexpr (HOUT) hout[row * 64 + col] = (_Float16)v;
                }
            }
        } else {
#pragma unroll
            for (int nt = 0; nt < 4; ++nt) {
                const int col = nt * 16 + ln;
#pragma unroll
                for (int r = 0; r < 8; ++r) {
                    const int row = tile * 16 + lh * 8 + r;
                    if (row < Nn) {
                        float v = acc[nt][r];
                        if constexpr (DINV) v *= dinv[row];
                        if constexpr (BIAS) v += bv[nt];
                        if constexpr (RELU) v = fmaxf(v, 0.0f);
                        ys[row * 64 + col] = v;
                        if constexpr (AGG)  agg[row * 64 + col] = v;
                        if constexpr (HOUT) hout[row * 64 + col] = (_Float16)v;
                    }
                }
            }
        }
    }
}

// ---------------------------------------------------------------------------
// Edge scatter: 16 lanes per edge; lane q adds ys[src][4q:4q+4] (b128 load)
// into agg[dst] with 4 f32 atomics. ys/agg are 25.6 MB each -> resident in
// the 192 MB L2, so the atomics stay on-chip.
// ---------------------------------------------------------------------------
__global__ __launch_bounds__(256) void k_scatter(
        const int* __restrict__ edges, const float* __restrict__ ys,
        float* __restrict__ agg, int E)
{
    const int gid = blockIdx.x * blockDim.x + threadIdx.x;
    const int e   = gid >> 4;
    const int q   = gid & 15;
    if (e >= E) return;
    const int s = edges[2 * e];
    const int d = edges[2 * e + 1];
    const float4 v = ((const float4*)(ys + (size_t)s * 64))[q];
    float* dst = agg + (size_t)d * 64 + q * 4;
    atomicAdd(dst + 0, v.x);
    atomicAdd(dst + 1, v.y);
    atomicAdd(dst + 2, v.z);
    atomicAdd(dst + 3, v.w);
}

// ---------------------------------------------------------------------------
// Finalize a conv layer: h = relu(dinv[dst] * agg + bias), stored as f16.
// (agg was initialized with the self-loop term ys in the GEMM epilogue.)
// ---------------------------------------------------------------------------
__global__ void k_finalize(const float* __restrict__ agg,
                           const float* __restrict__ dinv,
                           const float* __restrict__ bias,
                           _Float16* __restrict__ hout, int Nn)
{
    int gid = blockIdx.x * blockDim.x + threadIdx.x;
    if (gid >= Nn * 64) return;
    const int n = gid >> 6, f = gid & 63;
    float v = dinv[n] * agg[gid] + bias[f];
    hout[gid] = (_Float16)fmaxf(v, 0.0f);
}

// ---------------------------------------------------------------------------
// Head: sigmoid(concat(x, h5) @ Wl2 + bl2), done in full f32 for precision.
// ---------------------------------------------------------------------------
__global__ void k_head(const float* __restrict__ x, const float* __restrict__ h5,
                       const float* __restrict__ Wl2, const float* __restrict__ bl2,
                       float* __restrict__ out, int Nn)
{
    int n = blockIdx.x * blockDim.x + threadIdx.x;
    if (n >= Nn) return;
    float s = bl2[0];
#pragma unroll 4
    for (int f = 0; f < 32; ++f) s += x[n * 32 + f] * Wl2[f];
#pragma unroll 4
    for (int f = 0; f < 64; ++f) s += h5[n * 64 + f] * Wl2[32 + f];
    out[n] = 1.0f / (1.0f + expf(-s));
}

// ---------------------------------------------------------------------------
extern "C" void kernel_launch(void* const* d_in, const int* in_sizes, int n_in,
                              void* d_out, int out_size, void* d_ws, size_t ws_size,
                              hipStream_t stream)
{
    const float* x    = (const float*)d_in[0];
    const int*   edges= (const int*)  d_in[1];
    const float* W1   = (const float*)d_in[2];  const float* b1  = (const float*)d_in[3];
    const float* W2   = (const float*)d_in[4];  const float* b2  = (const float*)d_in[5];
    const float* W3   = (const float*)d_in[6];  const float* b3  = (const float*)d_in[7];
    const float* W4   = (const float*)d_in[8];  const float* b4  = (const float*)d_in[9];
    const float* Wl1  = (const float*)d_in[10]; const float* bl1 = (const float*)d_in[11];
    const float* Wl2  = (const float*)d_in[12]; const float* bl2 = (const float*)d_in[13];
    float* out = (float*)d_out;

    const int Nn = in_sizes[0] / 32;   // F = 32
    const int E  = in_sizes[1] / 2;    // edges are [E,2]
    const int Ntiles = (Nn + 15) / 16;

    // ---- carve workspace --------------------------------------------------
    char* ws = (char*)d_ws;
    size_t off = 0;
    auto carve = [&](size_t bytes) -> char* {
        char* p = ws + off;
        off += (bytes + 255) & ~(size_t)255;
        return p;
    };
    _Float16* xh   = (_Float16*)carve((size_t)Nn * 32 * 2);
    _Float16* h16  = (_Float16*)carve((size_t)Nn * 64 * 2);
    float*    ys   = (float*)   carve((size_t)Nn * 64 * 4);
    float*    agg  = (float*)   carve((size_t)Nn * 64 * 4);
    float*    h5   = (float*)   carve((size_t)Nn * 64 * 4);
    unsigned* deg  = (unsigned*)carve((size_t)Nn * 4);
    float*    dinv = (float*)   carve((size_t)Nn * 4);
    _Float16* w1h  = (_Float16*)carve(32 * 64 * 2);
    _Float16* w2h  = (_Float16*)carve(96 * 64 * 2);
    _Float16* w3h  = (_Float16*)carve(96 * 64 * 2);
    _Float16* w4h  = (_Float16*)carve(96 * 64 * 2);
    _Float16* wl1h = (_Float16*)carve(96 * 64 * 2);

    const int TB = 256;
    auto blocks = [](int n, int tb) { return (n + tb - 1) / tb; };

    // ---- precision conversions (per call; deterministic) ------------------
    k_f32_to_f16<<<blocks(Nn * 32, TB), TB, 0, stream>>>(x,   xh,   Nn * 32);
    k_f32_to_f16<<<blocks(32 * 64, TB), TB, 0, stream>>>(W1,  w1h,  32 * 64);
    k_f32_to_f16<<<blocks(96 * 64, TB), TB, 0, stream>>>(W2,  w2h,  96 * 64);
    k_f32_to_f16<<<blocks(96 * 64, TB), TB, 0, stream>>>(W3,  w3h,  96 * 64);
    k_f32_to_f16<<<blocks(96 * 64, TB), TB, 0, stream>>>(W4,  w4h,  96 * 64);
    k_f32_to_f16<<<blocks(96 * 64, TB), TB, 0, stream>>>(Wl1, wl1h, 96 * 64);

    // ---- degree / dinv ----------------------------------------------------
    k_deg_init <<<blocks(Nn, TB), TB, 0, stream>>>(deg, Nn);
    k_deg_count<<<blocks(E,  TB), TB, 0, stream>>>(edges, deg, E);
    k_dinv     <<<blocks(Nn, TB), TB, 0, stream>>>(deg, dinv, Nn);

    const int gemmBlocks    = blocks(Ntiles, 8);      // 8 waves/block, 1 tile/wave
    const int scatterBlocks = blocks(E * 16, TB);     // 16 lanes/edge
    const int finBlocks     = blocks(Nn * 64, TB);

    // ---- layer 1: x @ W1 --------------------------------------------------
    k_gcn_gemm<32, true, false, false, true, false><<<gemmBlocks, TB, 0, stream>>>(
        xh, (const _Float16*)nullptr, w1h, nullptr, dinv, ys, agg,
        (_Float16*)nullptr, Nn, Ntiles);
    k_scatter <<<scatterBlocks, TB, 0, stream>>>(edges, ys, agg, E);
    k_finalize<<<finBlocks, TB, 0, stream>>>(agg, dinv, b1, h16, Nn);

    // ---- layers 2..4: concat(x,h) @ W -------------------------------------
    const _Float16* Wh[3] = {w2h, w3h, w4h};
    const float*    Bs[3] = {b2, b3, b4};
    for (int l = 0; l < 3; ++l) {
        k_gcn_gemm<96, true, false, false, true, false><<<gemmBlocks, TB, 0, stream>>>(
            xh, h16, Wh[l], nullptr, dinv, ys, agg, (_Float16*)nullptr, Nn, Ntiles);
        k_scatter <<<scatterBlocks, TB, 0, stream>>>(edges, ys, agg, E);
        k_finalize<<<finBlocks, TB, 0, stream>>>(agg, dinv, Bs[l], h16, Nn);
    }

    // ---- MLP layer: h5 = relu(concat(x,h4) @ Wl1 + bl1) (no graph norm) ---
    k_gcn_gemm<96, false, true, true, false, false><<<gemmBlocks, TB, 0, stream>>>(
        xh, h16, wl1h, bl1, (const float*)nullptr, h5, (float*)nullptr,
        (_Float16*)nullptr, Nn, Ntiles);

    // ---- head: sigmoid(concat(x,h5) @ Wl2 + bl2), f32 ---------------------
    k_head<<<blocks(Nn, TB), TB, 0, stream>>>(x, h5, Wl2, bl2, out, Nn);
}
